// INN_Brain_44976897523927
// MI455X (gfx1250) — matrix operand order repaired and checked by reading, not compiled, not run on hardware
//
#include <hip/hip_runtime.h>
#include <hip/hip_bf16.h>
#include <math.h>

// Problem constants (from reference)
#define BB 8      // batch
#define SS 256    // sequence
#define VV 32000  // vocab
#define DD 512    // model dim
#define NN 16     // LSTM cells
#define RR 512    // cell hidden
#define HH 8      // heads
#define G4 2048   // 4*RR
#define DH 64     // DD/HH
#define MP 16     // padded M for WMMA tiles

typedef __bf16 bf16_t;
typedef __attribute__((ext_vector_type(16))) __bf16 v16bf;
typedef __attribute__((ext_vector_type(8)))  __bf16 v8bf;
typedef __attribute__((ext_vector_type(8)))  float  v8f;

static __device__ __forceinline__ bf16_t f2bf(float f) {
  unsigned u = __builtin_bit_cast(unsigned, f);
  unsigned r = u + 0x7fffu + ((u >> 16) & 1u);   // round-to-nearest-even
  unsigned short h = (unsigned short)(r >> 16);
  return __builtin_bit_cast(bf16_t, h);
}

// ---- WMMA operand loaders, exactly per CDNA5 ISA VGPR layout tables ----
// A (16x32 bf16, MxK): lane holds row M=lane&15; elems 0..7 = K[k0+8*(lane>>4) ..],
// elems 8..15 = same +16.
static __device__ __forceinline__ v16bf load_a16x32(const bf16_t* A, int lda, int k0, int lane) {
  int row  = lane & 15;
  int base = k0 + ((lane >> 4) << 3);
  const bf16_t* p = A + (size_t)row * lda + base;
  v8bf lo = *(const v8bf*)p;          // 16B aligned
  v8bf hi = *(const v8bf*)(p + 16);
  v16bf a;
#pragma unroll
  for (int i = 0; i < 8; ++i) { a[i] = lo[i]; a[i + 8] = hi[i]; }
  return a;
}
// B (32x16 bf16, KxN) sourced from row-major weight W[(out,in)] so column n of B
// is row n of W: lane holds col N=lane&15; elems 0..7 = K[k0+8*(lane>>4) ..], elems 8..15 = same +16.
static __device__ __forceinline__ v16bf load_b32x16(const bf16_t* W, int ldw, int n0, int k0, int lane) {
  int col  = n0 + (lane & 15);
  int base = k0 + ((lane >> 4) << 3);
  const bf16_t* p = W + (size_t)col * ldw + base;
  v8bf lo = *(const v8bf*)p;          // 16B aligned
  v8bf hi = *(const v8bf*)(p + 16);
  v16bf b;
#pragma unroll
  for (int i = 0; i < 8; ++i) { b[i] = lo[i]; b[i + 8] = hi[i]; }
  return b;
}
#define WMMA_BF16(acc, a, b) \
  (acc) = __builtin_amdgcn_wmma_f32_16x16x32_bf16(false, (a), false, (b), (short)0, (acc), false, false)

// ---------------- setup kernels ----------------
__global__ void k_cvt(const float* __restrict__ src, bf16_t* __restrict__ dst, int count) {
  int i = blockIdx.x * blockDim.x + threadIdx.x;
  if (i < count) dst[i] = f2bf(src[i]);
}

// Gather embeddings into (S, 16, D) padded, bf16 + f32 (rows 8..15 zero).
__global__ void k_gather(const int* __restrict__ x, const float* __restrict__ emb,
                         bf16_t* __restrict__ xsbf, float* __restrict__ xsf) {
  size_t idx = (size_t)blockIdx.x * blockDim.x + threadIdx.x;  // SS*MP*DD
  int t   = (int)(idx / (MP * DD));
  int rem = (int)(idx % (MP * DD));
  int row = rem / DD, d = rem % DD;
  float v = 0.f;
  if (row < BB) {
    int tok = x[row * SS + t];
    v = emb[(size_t)tok * DD + d];
  }
  xsbf[idx] = f2bf(v);
  xsf[idx]  = v;
}

// Initialize h/c state and zero the padded rows of staged activations.
__global__ void k_init(const float* __restrict__ h0, const float* __restrict__ c0,
                       float* __restrict__ hf, float* __restrict__ cf,
                       bf16_t* __restrict__ hbf, bf16_t* __restrict__ attbf,
                       bf16_t* __restrict__ mixedbf) {
  int idx = blockIdx.x * blockDim.x + threadIdx.x;  // NN*MP*RR
  if (idx >= NN * MP * RR) return;
  int n   = idx / (MP * RR);
  int rem = idx % (MP * RR);
  int row = rem / RR, r = rem % RR;
  float h = 0.f;
  if (row < BB) {
    int s = (n * BB + row) * RR + r;
    h = h0[s];
    hf[s] = h;
    cf[s] = c0[s];
  }
  hbf[idx] = f2bf(h);
  if (idx < MP * DD && (idx / DD) >= BB) {  // zero pad rows of att/mixed staging
    attbf[idx]   = f2bf(0.f);
    mixedbf[idx] = f2bf(0.f);
  }
}

// ---------------- per-step kernels ----------------
// gates[n,:,g] = xt @ W_ih[n]^T + h[n] @ W_hh[n]^T + (b_ih+b_hh); 2048 waves.
__global__ void k_gates(const bf16_t* __restrict__ xt, const bf16_t* __restrict__ hbf,
                        const bf16_t* __restrict__ wih, const bf16_t* __restrict__ whh,
                        const float* __restrict__ bih, const float* __restrict__ bhh,
                        float* __restrict__ gates) {
  int wave = (blockIdx.x * blockDim.x + threadIdx.x) >> 5;
  int lane = threadIdx.x & 31;
  int n  = wave >> 7;
  int n0 = (wave & 127) << 4;
  const bf16_t* Wi = wih + (size_t)n * G4 * DD;
  const bf16_t* Wh = whh + (size_t)n * G4 * RR;
  const bf16_t* Ah = hbf + (size_t)n * MP * RR;
  v8f acc = {};
#pragma unroll 2
  for (int k0 = 0; k0 < DD; k0 += 32) {
    v16bf a = load_a16x32(xt, DD, k0, lane);
    v16bf b = load_b32x16(Wi, DD, n0, k0, lane);
    WMMA_BF16(acc, a, b);
  }
#pragma unroll 2
  for (int k0 = 0; k0 < RR; k0 += 32) {
    v16bf a = load_a16x32(Ah, RR, k0, lane);
    v16bf b = load_b32x16(Wh, RR, n0, k0, lane);
    WMMA_BF16(acc, a, b);
  }
  int nc = lane & 15, mh = (lane >> 4) << 3;
  float bias = bih[n * G4 + n0 + nc] + bhh[n * G4 + n0 + nc];
#pragma unroll
  for (int r = 0; r < 8; ++r) {
    int m = r + mh;
    gates[((size_t)(n * MP + m)) * G4 + n0 + nc] = acc[r] + bias;
  }
}

__global__ void k_lstm(const float* __restrict__ gates, float* __restrict__ hf,
                       float* __restrict__ cf, bf16_t* __restrict__ hbf) {
  int idx = blockIdx.x * blockDim.x + threadIdx.x;  // NN*BB*RR
  if (idx >= NN * BB * RR) return;
  int n   = idx / (BB * RR);
  int rem = idx % (BB * RR);
  int b = rem / RR, r = rem % RR;
  const float* g = gates + ((size_t)(n * MP + b)) * G4;
  float ig = 1.f / (1.f + __expf(-g[r]));
  float fg = 1.f / (1.f + __expf(-g[RR + r]));
  float gg = tanhf(g[2 * RR + r]);
  float og = 1.f / (1.f + __expf(-g[3 * RR + r]));
  float c = fg * cf[idx] + ig * gg;
  float h = og * tanhf(c);
  cf[idx] = c;
  hf[idx] = h;
  hbf[((size_t)n * MP + b) * RR + r] = f2bf(h);
}

// nout[b,n,:] = h[n,b,:] @ W_np[n]^T + b_np[n]; stored bf16 as 128 x D. 512 waves.
__global__ void k_nout(const bf16_t* __restrict__ hbf, const bf16_t* __restrict__ wnp,
                       const float* __restrict__ bnp, bf16_t* __restrict__ noutbf) {
  int wave = (blockIdx.x * blockDim.x + threadIdx.x) >> 5;
  int lane = threadIdx.x & 31;
  int n  = wave >> 5;
  int n0 = (wave & 31) << 4;
  const bf16_t* A  = hbf + (size_t)n * MP * RR;
  const bf16_t* Wn = wnp + (size_t)n * DD * RR;
  v8f acc = {};
#pragma unroll 2
  for (int k0 = 0; k0 < RR; k0 += 32) {
    v16bf a = load_a16x32(A, RR, k0, lane);
    v16bf b = load_b32x16(Wn, RR, n0, k0, lane);
    WMMA_BF16(acc, a, b);
  }
  int nc = lane & 15, mh = (lane >> 4) << 3;
  float bias = bnp[n * DD + n0 + nc];
#pragma unroll
  for (int r = 0; r < 8; ++r) {
    int m = r + mh;
    if (m < BB) noutbf[((size_t)(m * NN + n)) * DD + n0 + nc] = f2bf(acc[r] + bias);
  }
}

// q = xt @ Wq^T + bq ; 32 waves.
__global__ void k_q(const bf16_t* __restrict__ xt, const bf16_t* __restrict__ inwbf,
                    const float* __restrict__ inb, float* __restrict__ q) {
  int wave = (blockIdx.x * blockDim.x + threadIdx.x) >> 5;
  int lane = threadIdx.x & 31;
  int n0 = wave << 4;
  v8f acc = {};
#pragma unroll 2
  for (int k0 = 0; k0 < DD; k0 += 32) {
    v16bf a = load_a16x32(xt, DD, k0, lane);
    v16bf b = load_b32x16(inwbf, DD, n0, k0, lane);
    WMMA_BF16(acc, a, b);
  }
  int nc = lane & 15, mh = (lane >> 4) << 3;
  float bias = inb[n0 + nc];
#pragma unroll
  for (int r = 0; r < 8; ++r) {
    int m = r + mh;
    if (m < BB) q[(size_t)m * DD + n0 + nc] = acc[r] + bias;
  }
}

// k/v = nout(128xD) @ Wk^T / Wv^T ; M=128 (8 tiles) x N=512 (32 tiles) x 2 = 512 waves.
__global__ void k_kv(const bf16_t* __restrict__ noutbf, const bf16_t* __restrict__ inwbf,
                     const float* __restrict__ inb, float* __restrict__ kbuf,
                     float* __restrict__ vbuf) {
  int wave = (blockIdx.x * blockDim.x + threadIdx.x) >> 5;
  int lane = threadIdx.x & 31;
  int which = wave >> 8;            // 0=k, 1=v
  int mt    = (wave >> 5) & 7;
  int n0    = (wave & 31) << 4;
  const bf16_t* A = noutbf + (size_t)mt * 16 * DD;
  const bf16_t* W = inwbf + (size_t)(DD + which * DD) * DD;
  const float*  bi = inb + DD + which * DD;
  float* dst = which ? vbuf : kbuf;
  v8f acc = {};
#pragma unroll 2
  for (int k0 = 0; k0 < DD; k0 += 32) {
    v16bf a = load_a16x32(A, DD, k0, lane);
    v16bf b = load_b32x16(W, DD, n0, k0, lane);
    WMMA_BF16(acc, a, b);
  }
  int nc = lane & 15, mh = (lane >> 4) << 3;
  float bias = bi[n0 + nc];
#pragma unroll
  for (int r = 0; r < 8; ++r) {
    int m = r + mh;
    dst[((size_t)(mt * 16 + m)) * DD + n0 + nc] = acc[r] + bias;
  }
}

// attention over NN=16 slots; one wave per (b,h); lane owns elements e and e+32 of dh=64.
__global__ void k_attn(const float* __restrict__ q, const float* __restrict__ kb,
                       const float* __restrict__ vb, bf16_t* __restrict__ attbf) {
  int wave = (blockIdx.x * blockDim.x + threadIdx.x) >> 5;
  int lane = threadIdx.x & 31;
  int b = wave >> 3, h = wave & 7;
  int e0 = lane, e1 = lane + 32;
  float q0 = q[(size_t)b * DD + h * DH + e0];
  float q1 = q[(size_t)b * DD + h * DH + e1];
  float sc[NN];
#pragma unroll
  for (int n = 0; n < NN; ++n) {
    const float* kp = kb + ((size_t)(b * NN + n)) * DD + h * DH;
    float p = q0 * kp[e0] + q1 * kp[e1];
#pragma unroll
    for (int off = 16; off; off >>= 1) p += __shfl_xor(p, off, 32);
    sc[n] = p * 0.125f;  // 1/sqrt(64)
  }
  float mx = sc[0];
#pragma unroll
  for (int n = 1; n < NN; ++n) mx = fmaxf(mx, sc[n]);
  float s = 0.f;
#pragma unroll
  for (int n = 0; n < NN; ++n) { sc[n] = __expf(sc[n] - mx); s += sc[n]; }
  float inv = 1.f / s, a0 = 0.f, a1 = 0.f;
#pragma unroll
  for (int n = 0; n < NN; ++n) {
    const float* vp = vb + ((size_t)(b * NN + n)) * DD + h * DH;
    float w = sc[n] * inv;
    a0 += w * vp[e0];
    a1 += w * vp[e1];
  }
  attbf[(size_t)b * DD + h * DH + e0] = f2bf(a0);
  attbf[(size_t)b * DD + h * DH + e1] = f2bf(a1);
}

// y = att @ op_w^T + op_b + xt (residual, f32); 32 waves.
__global__ void k_op(const bf16_t* __restrict__ attbf, const bf16_t* __restrict__ opwbf,
                     const float* __restrict__ opb, const float* __restrict__ xtf,
                     float* __restrict__ y) {
  int wave = (blockIdx.x * blockDim.x + threadIdx.x) >> 5;
  int lane = threadIdx.x & 31;
  int n0 = wave << 4;
  v8f acc = {};
#pragma unroll 2
  for (int k0 = 0; k0 < DD; k0 += 32) {
    v16bf a = load_a16x32(attbf, DD, k0, lane);
    v16bf b = load_b32x16(opwbf, DD, n0, k0, lane);
    WMMA_BF16(acc, a, b);
  }
  int nc = lane & 15, mh = (lane >> 4) << 3;
  float bias = opb[n0 + nc];
#pragma unroll
  for (int r = 0; r < 8; ++r) {
    int m = r + mh;
    if (m < BB) y[(size_t)m * DD + n0 + nc] = acc[r] + bias + xtf[(size_t)m * DD + n0 + nc];
  }
}

// LayerNorm over D per batch row -> mixed bf16 (padded rows stay zero).
__global__ void k_ln(const float* __restrict__ y, const float* __restrict__ g,
                     const float* __restrict__ be, bf16_t* __restrict__ mixedbf) {
  __shared__ float red[256];
  int b = blockIdx.x, tid = threadIdx.x;
  const float* yr = y + (size_t)b * DD;
  float v0 = yr[tid], v1 = yr[tid + 256];
  red[tid] = v0 + v1;
  __syncthreads();
  for (int off = 128; off; off >>= 1) { if (tid < off) red[tid] += red[tid + off]; __syncthreads(); }
  float mu = red[0] * (1.f / DD);
  __syncthreads();
  float d0 = v0 - mu, d1 = v1 - mu;
  red[tid] = d0 * d0 + d1 * d1;
  __syncthreads();
  for (int off = 128; off; off >>= 1) { if (tid < off) red[tid] += red[tid + off]; __syncthreads(); }
  float rstd = rsqrtf(red[0] * (1.f / DD) + 1e-5f);
  mixedbf[(size_t)b * DD + tid]       = f2bf(d0 * rstd * g[tid] + be[tid]);
  mixedbf[(size_t)b * DD + tid + 256] = f2bf(d1 * rstd * g[tid + 256] + be[tid + 256]);
}

// logits = mixed @ head_w^T + head_b -> d_out[(b,t,:)]; 2000 waves; L2-streamed head_w.
__global__ void k_head(const bf16_t* __restrict__ mixedbf, const bf16_t* __restrict__ hwbf,
                       const float* __restrict__ hb, float* __restrict__ out, int t) {
  int wave = (blockIdx.x * blockDim.x + threadIdx.x) >> 5;
  int lane = threadIdx.x & 31;
  int n0 = wave << 4;
  const bf16_t* Wrow = hwbf + (size_t)(n0 + (lane & 15)) * DD;
  v8f acc = {};
#pragma unroll 2
  for (int k0 = 0; k0 < DD; k0 += 32) {
    if (k0 + 64 < DD) __builtin_prefetch(Wrow + k0 + 64, 0, 1);  // global_prefetch_b8
    v16bf a = load_a16x32(mixedbf, DD, k0, lane);
    v16bf b = load_b32x16(hwbf, DD, n0, k0, lane);
    WMMA_BF16(acc, a, b);
  }
  int nc = lane & 15, mh = (lane >> 4) << 3;
  float bias = hb[n0 + nc];
#pragma unroll
  for (int r = 0; r < 8; ++r) {
    int m = r + mh;
    if (m < BB) out[((size_t)m * SS + t) * VV + n0 + nc] = acc[r] + bias;
  }
}

__global__ void k_final(const float* __restrict__ hf, const float* __restrict__ cf,
                        float* __restrict__ out) {
  int idx = blockIdx.x * blockDim.x + threadIdx.x;
  if (idx >= NN * BB * RR) return;
  size_t base = (size_t)BB * SS * VV;
  out[base + idx] = hf[idx];
  out[base + NN * BB * RR + idx] = cf[idx];
}

extern "C" void kernel_launch(void* const* d_in, const int* in_sizes, int n_in,
                              void* d_out, int out_size, void* d_ws, size_t ws_size,
                              hipStream_t stream) {
  const int*   x    = (const int*)d_in[0];
  const float* h0   = (const float*)d_in[1];
  const float* c0   = (const float*)d_in[2];
  const float* emb  = (const float*)d_in[3];
  const float* Wih  = (const float*)d_in[4];
  const float* bih  = (const float*)d_in[5];
  const float* Whh  = (const float*)d_in[6];
  const float* bhh  = (const float*)d_in[7];
  const float* Wnp  = (const float*)d_in[8];
  const float* bnp  = (const float*)d_in[9];
  const float* inw  = (const float*)d_in[10];
  const float* inb  = (const float*)d_in[11];
  const float* opw  = (const float*)d_in[12];
  const float* opb  = (const float*)d_in[13];
  const float* lng  = (const float*)d_in[14];
  const float* lnb  = (const float*)d_in[15];
  const float* hw   = (const float*)d_in[16];
  const float* hb   = (const float*)d_in[17];
  float* out = (float*)d_out;

  // workspace carve-up (256B aligned)
  char* base = (char*)d_ws;
  size_t off = 0;
  auto carve = [&](size_t bytes) -> char* {
    off = (off + 255) & ~(size_t)255;
    char* p = base + off;
    off += bytes;
    return p;
  };
  bf16_t* hwbf   = (bf16_t*)carve((size_t)VV * DD * 2);
  bf16_t* wihbf  = (bf16_t*)carve((size_t)NN * G4 * DD * 2);
  bf16_t* whhbf  = (bf16_t*)carve((size_t)NN * G4 * RR * 2);
  bf16_t* wnpbf  = (bf16_t*)carve((size_t)NN * DD * RR * 2);
  bf16_t* inwbf  = (bf16_t*)carve((size_t)3 * DD * DD * 2);
  bf16_t* opwbf  = (bf16_t*)carve((size_t)DD * DD * 2);
  bf16_t* xsbf   = (bf16_t*)carve((size_t)SS * MP * DD * 2);
  float*  xsf    = (float*)carve((size_t)SS * MP * DD * 4);
  bf16_t* hbf    = (bf16_t*)carve((size_t)NN * MP * RR * 2);
  float*  hfst   = (float*)carve((size_t)NN * BB * RR * 4);
  float*  cfst   = (float*)carve((size_t)NN * BB * RR * 4);
  float*  gates  = (float*)carve((size_t)NN * MP * G4 * 4);
  bf16_t* noutbf = (bf16_t*)carve((size_t)BB * NN * DD * 2);
  float*  qbuf   = (float*)carve((size_t)MP * DD * 4);
  float*  kbuf   = (float*)carve((size_t)BB * NN * DD * 4);
  float*  vbuf   = (float*)carve((size_t)BB * NN * DD * 4);
  bf16_t* attbf  = (bf16_t*)carve((size_t)MP * DD * 2);
  float*  ybuf   = (float*)carve((size_t)MP * DD * 4);
  bf16_t* mixbf  = (bf16_t*)carve((size_t)MP * DD * 2);
  (void)ws_size; (void)n_in; (void)in_sizes; (void)out_size;

  // one-time f32 -> bf16 weight conversion (fits L2 afterwards)
  auto cvt = [&](const float* s, bf16_t* d, size_t n) {
    k_cvt<<<(unsigned)((n + 255) / 256), 256, 0, stream>>>(s, d, (int)n);
  };
  cvt(hw,  hwbf,  (size_t)VV * DD);
  cvt(Wih, wihbf, (size_t)NN * G4 * DD);
  cvt(Whh, whhbf, (size_t)NN * G4 * RR);
  cvt(Wnp, wnpbf, (size_t)NN * DD * RR);
  cvt(inw, inwbf, (size_t)3 * DD * DD);
  cvt(opw, opwbf, (size_t)DD * DD);

  k_gather<<<(SS * MP * DD) / 256, 256, 0, stream>>>(x, emb, xsbf, xsf);
  k_init<<<(NN * MP * RR) / 256, 256, 0, stream>>>(h0, c0, hfst, cfst, hbf, attbf, mixbf);

  for (int t = 0; t < SS; ++t) {
    const bf16_t* xt  = xsbf + (size_t)t * MP * DD;
    const float*  xtf = xsf  + (size_t)t * MP * DD;
    k_gates<<<256, 256, 0, stream>>>(xt, hbf, wihbf, whhbf, bih, bhh, gates);
    k_lstm<<<256, 256, 0, stream>>>(gates, hfst, cfst, hbf);
    k_nout<<<64, 256, 0, stream>>>(hbf, wnpbf, bnp, noutbf);
    k_q<<<4, 256, 0, stream>>>(xt, inwbf, inb, qbuf);
    k_kv<<<64, 256, 0, stream>>>(noutbf, inwbf, inb, kbuf, vbuf);
    k_attn<<<8, 256, 0, stream>>>(qbuf, kbuf, vbuf, attbf);
    k_op<<<4, 256, 0, stream>>>(attbf, opwbf, opb, xtf, ybuf);
    k_ln<<<BB, 256, 0, stream>>>(ybuf, lng, lnb, mixbf);
    k_head<<<250, 256, 0, stream>>>(mixbf, hwbf, hb, out, t);
  }
  k_final<<<(NN * BB * RR) / 256, 256, 0, stream>>>(hfst, cfst, out);
}
